// Joiner_14035953123456
// MI455X (gfx1250) — compile-verified, matrix operand order checked by reading, not run on hardware
//
#include <hip/hip_runtime.h>
#include <hip/hip_bf16.h>
#include <math.h>

// ---------------- types ----------------
typedef __attribute__((ext_vector_type(16))) __bf16 v16bf;
typedef __attribute__((ext_vector_type(8)))  float  v8f;
typedef __attribute__((ext_vector_type(4)))  __bf16 v4bf;

#define N_BATCH 8
#define T_DIM   200
#define U_DIM   50
#define J_DIM   512
#define VOCAB   500
#define RT_ROWS (T_DIM * U_DIM)   // 10000 (t,u) rows per batch

union FragBF { v16bf v; uint4 q[2]; };

__device__ __forceinline__ float4 ld4f(const float* p) { return *(const float4*)p; }

// async global -> LDS copy of 16B per lane (CDNA5 path, tracked by ASYNCcnt)
#define ASYNC_LDS_B128(lds32, gaddr64, OFF)                                    \
    asm volatile("global_load_async_to_lds_b128 %0, %1, off offset:" #OFF     \
                 :: "v"(lds32), "v"(gaddr64) : "memory")

__device__ __forceinline__ void wait_asynccnt0() {
    asm volatile("s_wait_asynccnt 0x0" ::: "memory");
}

// ---------------- conversion kernels ----------------
__global__ void cvt_f32_bf16_k(const float* __restrict__ in, __bf16* __restrict__ out, int n) {
    int i = blockIdx.x * blockDim.x + threadIdx.x;
    if (i < n) out[i] = (__bf16)in[i];
}

// W_out (500,512) f32 -> (512,512) bf16, zero-padded rows
__global__ void cvt_wout_k(const float* __restrict__ in, __bf16* __restrict__ out) {
    int i = blockIdx.x * blockDim.x + threadIdx.x;   // 0 .. 512*512-1
    if (i < 512 * 512) {
        int r = i >> 9;
        out[i] = (r < VOCAB) ? (__bf16)in[i] : (__bf16)0.0f;
    }
}

// ---------------- WMMA projection GEMM ----------------
// out[m][j] = sum_k A[m][k] * W[j][k] + bias[j];  A:(M,512) bf16, W:(512,512) bf16, out f32
__global__ __launch_bounds__(256)
void proj_wmma_kernel(const __bf16* __restrict__ A,
                      const __bf16* __restrict__ W,
                      const float*  __restrict__ bias,
                      float*        __restrict__ out,
                      int M)
{
    __shared__ __align__(16) __bf16 xt[32][520];  // 32 rows x 512 k (padded)
    __shared__ __align__(16) __bf16 wt[256][40];  // 256 cols x 32 k (padded)

    const int tid = threadIdx.x;
    const int r0  = blockIdx.x * 32;

    // stage 32 rows of A
    {
        int row = tid >> 3;
        int k0  = (tid & 7) * 64;
        int r   = r0 + row;
        int re  = (r < M) ? r : (M - 1);
        const uint4* src = (const uint4*)(A + (size_t)re * J_DIM + k0);
        uint4* dst = (uint4*)&xt[row][k0];
#pragma unroll
        for (int q = 0; q < 8; ++q) dst[q] = src[q];
    }

    const int w        = tid >> 5;
    const int lane     = tid & 31;
    const int h        = lane >> 4;
    const int nl       = lane & 15;
    const int rowStrip = (w & 1) * 16;
    const int colGrp   = (w >> 1) * 64;
    const v8f vzero = {0.f,0.f,0.f,0.f,0.f,0.f,0.f,0.f};

    for (int pass = 0; pass < 2; ++pass) {
        v8f acc[4];
#pragma unroll
        for (int ft = 0; ft < 4; ++ft) acc[ft] = vzero;

        for (int kc = 0; kc < J_DIM; kc += 32) {
            __syncthreads();
            {   // stage W tile: cols [pass*256, +256), k [kc, kc+32)
                const uint4* src = (const uint4*)(W + (size_t)(pass * 256 + tid) * J_DIM + kc);
                uint4* dst = (uint4*)&wt[tid][0];
                dst[0] = src[0]; dst[1] = src[1]; dst[2] = src[2]; dst[3] = src[3];
            }
            __syncthreads();

            FragBF a;
            a.q[0] = *(const uint4*)&xt[rowStrip + nl][kc + 8 * h];
            a.q[1] = *(const uint4*)&xt[rowStrip + nl][kc + 16 + 8 * h];
#pragma unroll
            for (int ft = 0; ft < 4; ++ft) {
                FragBF b;
                int c = colGrp + ft * 16 + nl;
                b.q[0] = *(const uint4*)&wt[c][8 * h];
                b.q[1] = *(const uint4*)&wt[c][16 + 8 * h];
                acc[ft] = __builtin_amdgcn_wmma_f32_16x16x32_bf16(
                    false, a.v, false, b.v, (short)0, acc[ft], false, false);
            }
        }

        // epilogue
#pragma unroll
        for (int ft = 0; ft < 4; ++ft) {
            int col = pass * 256 + colGrp + ft * 16 + nl;
            float bv = bias[col];
#pragma unroll
            for (int i = 0; i < 8; ++i) {
                int r = r0 + rowStrip + 8 * h + i;
                if (r < M)
                    out[(size_t)r * J_DIM + col] = acc[ft][i] + bv;
            }
        }
    }
}

// ---------------- fused joiner: tanh(enc+dec) @ W_out^T + b_out ----------------
// Per workgroup: 32 (t,u) rows x 512 cols. Each wave: 32 rows x 64 cols
// (2 A-frags x 4 B-frags -> 8 WMMAs, every B frag reused twice).
// X tile staged once (tanh once per element); W_out k-slices double-buffered in
// LDS and fetched with async global->LDS (ASYNCcnt), one barrier per k-chunk.
__global__ __launch_bounds__(256)
void joiner_wmma_kernel(const float* __restrict__ encP,   // (N*T, 512) f32
                        const float* __restrict__ decP,   // (N*U, 512) f32
                        const __bf16* __restrict__ WoutB, // (512, 512) bf16
                        const float* __restrict__ b_out,  // (500) f32
                        float* __restrict__ out)          // (N, T, U, 500) f32
{
    __shared__ __align__(16) __bf16 xt[32][520];     // 32 rows x 512 k   33.3KB
    __shared__ __align__(16) __bf16 wt[2][512][40];  // 2 x (512c x 32k)  81.9KB

    const int tid = threadIdx.x;
    const int r0  = blockIdx.x * 32;   // (t,u) flat row block
    const int nb  = blockIdx.y;        // batch

    const int w    = tid >> 5;         // wave 0..7 -> col group of 64
    const int lane = tid & 31;
    const int h    = lane >> 4;
    const int nl   = lane & 15;

    // ---- stage X = bf16(tanh(enc+dec)) : 32 rows x 512 k, computed ONCE ----
    {
        int srow = tid >> 3;
        int sk0  = (tid & 7) * 64;
        int r    = r0 + srow;
        int re   = (r < RT_ROWS) ? r : (RT_ROWS - 1);
        int t = re / U_DIM, u = re % U_DIM;
        const float* ep = encP + ((size_t)(nb * T_DIM + t)) * J_DIM + sk0;
        const float* dp = decP + ((size_t)(nb * U_DIM + u)) * J_DIM + sk0;
#pragma unroll 4
        for (int kk = 0; kk < 64; kk += 4) {
            float4 e = ld4f(ep + kk);
            float4 d = ld4f(dp + kk);
            v4bf v;
            v.x = (__bf16)tanhf(e.x + d.x);
            v.y = (__bf16)tanhf(e.y + d.y);
            v.z = (__bf16)tanhf(e.z + d.z);
            v.w = (__bf16)tanhf(e.w + d.w);
            *(v4bf*)&xt[srow][sk0 + kk] = v;
        }
    }

    // per-thread staging addresses: 2 cols (tid, tid+256) x 64B per k-chunk
    const __bf16* g0p = WoutB + (size_t)tid * J_DIM;           // col tid
    const __bf16* g1p = g0p + (size_t)256 * J_DIM;             // col tid+256
    const unsigned l0a = (unsigned)(size_t)&wt[0][tid][0];
    const unsigned l1a = (unsigned)(size_t)&wt[0][tid + 256][0];
    const unsigned lstep = (unsigned)((size_t)&wt[1][0][0] - (size_t)&wt[0][0][0]);

    // prologue: async-fill buffer 0 with k-chunk 0
    {
        unsigned long long g0 = (unsigned long long)(size_t)g0p;
        unsigned long long g1 = (unsigned long long)(size_t)g1p;
        ASYNC_LDS_B128(l0a, g0, 0);  ASYNC_LDS_B128(l0a, g0, 16);
        ASYNC_LDS_B128(l0a, g0, 32); ASYNC_LDS_B128(l0a, g0, 48);
        ASYNC_LDS_B128(l1a, g1, 0);  ASYNC_LDS_B128(l1a, g1, 16);
        ASYNC_LDS_B128(l1a, g1, 32); ASYNC_LDS_B128(l1a, g1, 48);
    }

    v8f acc[2][4];
    const v8f vzero = {0.f,0.f,0.f,0.f,0.f,0.f,0.f,0.f};
#pragma unroll
    for (int rs = 0; rs < 2; ++rs)
#pragma unroll
        for (int ft = 0; ft < 4; ++ft) acc[rs][ft] = vzero;

    for (int ci = 0; ci < 16; ++ci) {          // 16 k-chunks of 32
        const int buf = ci & 1;
        wait_asynccnt0();                       // own async loads for `buf` done
        __syncthreads();                        // everyone's loads done; prev reads done

        if (ci + 1 < 16) {                      // async-fill other buffer with next chunk
            unsigned nl0 = l0a + (1 - buf) * lstep;
            unsigned nl1 = l1a + (1 - buf) * lstep;
            unsigned long long g0 = (unsigned long long)(size_t)(g0p + (ci + 1) * 32);
            unsigned long long g1 = (unsigned long long)(size_t)(g1p + (ci + 1) * 32);
            ASYNC_LDS_B128(nl0, g0, 0);  ASYNC_LDS_B128(nl0, g0, 16);
            ASYNC_LDS_B128(nl0, g0, 32); ASYNC_LDS_B128(nl0, g0, 48);
            ASYNC_LDS_B128(nl1, g1, 0);  ASYNC_LDS_B128(nl1, g1, 16);
            ASYNC_LDS_B128(nl1, g1, 32); ASYNC_LDS_B128(nl1, g1, 48);
        }

        const int kc = ci * 32;
        FragBF a0, a1;
        a0.q[0] = *(const uint4*)&xt[nl][kc + 8 * h];
        a0.q[1] = *(const uint4*)&xt[nl][kc + 16 + 8 * h];
        a1.q[0] = *(const uint4*)&xt[16 + nl][kc + 8 * h];
        a1.q[1] = *(const uint4*)&xt[16 + nl][kc + 16 + 8 * h];
#pragma unroll
        for (int ft = 0; ft < 4; ++ft) {
            FragBF b;
            int c = w * 64 + ft * 16 + nl;
            b.q[0] = *(const uint4*)&wt[buf][c][8 * h];
            b.q[1] = *(const uint4*)&wt[buf][c][16 + 8 * h];
            acc[0][ft] = __builtin_amdgcn_wmma_f32_16x16x32_bf16(
                false, a0.v, false, b.v, (short)0, acc[0][ft], false, false);
            acc[1][ft] = __builtin_amdgcn_wmma_f32_16x16x32_bf16(
                false, a1.v, false, b.v, (short)0, acc[1][ft], false, false);
        }
    }

    // epilogue: bias + masked store (32 rows x 64 cols per wave)
#pragma unroll
    for (int rs = 0; rs < 2; ++rs) {
#pragma unroll
        for (int ft = 0; ft < 4; ++ft) {
            int col = w * 64 + ft * 16 + nl;
            if (col < VOCAB) {
                float bv = b_out[col];
#pragma unroll
                for (int i = 0; i < 8; ++i) {
                    int rr = r0 + rs * 16 + 8 * h + i;
                    if (rr < RT_ROWS)
                        out[((size_t)nb * RT_ROWS + rr) * VOCAB + col] = acc[rs][ft][i] + bv;
                }
            }
        }
    }
}

// ---------------- launcher ----------------
extern "C" void kernel_launch(void* const* d_in, const int* in_sizes, int n_in,
                              void* d_out, int out_size, void* d_ws, size_t ws_size,
                              hipStream_t stream)
{
    const float* enc_in = (const float*)d_in[0];   // (8,200,512)
    const float* dec_in = (const float*)d_in[1];   // (8,50,512)
    const float* W_enc  = (const float*)d_in[2];   // (512,512)
    const float* b_enc  = (const float*)d_in[3];   // (512)
    const float* W_dec  = (const float*)d_in[4];   // (512,512)
    const float* b_dec  = (const float*)d_in[5];   // (512)
    const float* W_out  = (const float*)d_in[6];   // (500,512)
    const float* b_out  = (const float*)d_in[7];   // (500)
    float* out = (float*)d_out;

    const int M_ENC = N_BATCH * T_DIM;   // 1600
    const int M_DEC = N_BATCH * U_DIM;   // 400

    char* ws = (char*)d_ws;
    size_t off = 0;
    __bf16* encB  = (__bf16*)(ws + off); off += (size_t)M_ENC * J_DIM * 2;
    __bf16* decB  = (__bf16*)(ws + off); off += (size_t)M_DEC * J_DIM * 2;
    __bf16* WencB = (__bf16*)(ws + off); off += (size_t)J_DIM * J_DIM * 2;
    __bf16* WdecB = (__bf16*)(ws + off); off += (size_t)J_DIM * J_DIM * 2;
    __bf16* WoutB = (__bf16*)(ws + off); off += (size_t)J_DIM * J_DIM * 2;
    float*  encP  = (float*)(ws + off);  off += (size_t)M_ENC * J_DIM * 4;
    float*  decP  = (float*)(ws + off);  off += (size_t)M_DEC * J_DIM * 4;
    (void)ws_size; (void)in_sizes; (void)n_in; (void)out_size;

    // 1) precision conversion (bf16 operands for WMMA)
    cvt_f32_bf16_k<<<(M_ENC * J_DIM + 255) / 256, 256, 0, stream>>>(enc_in, encB, M_ENC * J_DIM);
    cvt_f32_bf16_k<<<(M_DEC * J_DIM + 255) / 256, 256, 0, stream>>>(dec_in, decB, M_DEC * J_DIM);
    cvt_f32_bf16_k<<<(J_DIM * J_DIM + 255) / 256, 256, 0, stream>>>(W_enc, WencB, J_DIM * J_DIM);
    cvt_f32_bf16_k<<<(J_DIM * J_DIM + 255) / 256, 256, 0, stream>>>(W_dec, WdecB, J_DIM * J_DIM);
    cvt_wout_k<<<(512 * 512) / 256, 256, 0, stream>>>(W_out, WoutB);

    // 2) input projections (WMMA GEMMs, f32 accum out)
    proj_wmma_kernel<<<dim3((M_ENC + 31) / 32), 256, 0, stream>>>(encB, WencB, b_enc, encP, M_ENC);
    proj_wmma_kernel<<<dim3((M_DEC + 31) / 32), 256, 0, stream>>>(decB, WdecB, b_dec, decP, M_DEC);

    // 3) fused tanh-broadcast + output GEMM (async double-buffered W staging)
    joiner_wmma_kernel<<<dim3((RT_ROWS + 31) / 32, N_BATCH), 256, 0, stream>>>(
        encP, decP, WoutB, b_out, out);
}
